// TrackEdgeGNN_31224412242360
// MI455X (gfx1250) — matrix operand order repaired and checked by reading, not compiled, not run on hardware
//
#include <hip/hip_runtime.h>

// =============================== types ====================================
typedef __attribute__((ext_vector_type(16))) __bf16         v16bf;
typedef __attribute__((ext_vector_type(8)))  float          v8f;
typedef __attribute__((ext_vector_type(8)))  unsigned short v8us;
typedef __attribute__((ext_vector_type(16))) unsigned short v16us;

struct SegDesc { const float* base; const int* idx; int width; };

// ============================ device helpers ==============================
static __device__ __forceinline__ unsigned short f2bf_bits(float f) {
    unsigned u = __float_as_uint(f);
    unsigned r = u + 0x7fffu + ((u >> 16) & 1u);   // round-to-nearest-even
    return (unsigned short)(r >> 16);
}
static __device__ __forceinline__ v16bf mk16(v8us lo, v8us hi) {
    v16us u = __builtin_shufflevector(lo, hi, 0,1,2,3,4,5,6,7,8,9,10,11,12,13,14,15);
    return __builtin_bit_cast(v16bf, u);
}
// monotonic float <-> uint mapping for atomicMax on floats
static __device__ __forceinline__ unsigned ford_enc(float f) {
    unsigned u = __float_as_uint(f);
    return (u & 0x80000000u) ? ~u : (u | 0x80000000u);
}
static __device__ __forceinline__ float ford_dec(unsigned u) {
    return (u & 0x80000000u) ? __uint_as_float(u ^ 0x80000000u)
                             : __uint_as_float(~u);
}

// ========================= generic WMMA GEMM ==============================
// C[rows, Nout] = act(A[rows,K] @ W[K,Nout] + bias) (+res); A gathered from up
// to 3 segments (fused concat + row gather). Optional alpha-weighted atomic
// scatter epilogue (fused attention segment_sum).
// Block: 256 thr = 8 waves = 2 row-tiles x 4 col-tiles => 32 rows x 64 cols.
// A tile (32 x Kpad bf16, Kpad<=576) staged once in LDS, branch-free; B read
// from pre-packed fragment-order weights (2x16B vector loads per WMMA).
__global__ __launch_bounds__(256) void k_gemm(
    SegDesc s0, SegDesc s1, SegDesc s2,
    int rows, int K,
    const unsigned short* __restrict__ Wp,    // packed bf16 [Kpad/32][Npad][32]
    const float* __restrict__ bias,           // [Nout] or null
    int Nout,
    float* __restrict__ out,                  // [rows, Nout] (unused if scatter)
    const float* __restrict__ res,            // optional residual [rows, Nout]
    int act,                                  // 0 none, 1 relu
    int scatter,
    const float* __restrict__ alpha,          // [rows, 4]
    const int* __restrict__ dstIdx,           // [rows]
    float* __restrict__ xacc)                 // [Nnodes, 256]
{
    const int Kpad = (K + 63) & ~63;          // multiple of 64 (<= 576 here)
    const int Npad = (Nout + 15) & ~15;

    __shared__ __align__(16) unsigned short atile[32 * 576];
    __shared__ const float*                 rowp[3][32];

    const int tid  = threadIdx.x;
    const int lane = tid & 31;                // wave32
    const int wid  = tid >> 5;
    const int rowBase = blockIdx.x * 32 + (wid >> 2) * 16;
    const int ncol0   = blockIdx.y * 64 + (wid & 3) * 16;
    const int col     = ncol0 + (lane & 15);

    // -------- resolve per-row segment base pointers (once) --------
    if (tid < 96) {
        int seg = tid >> 5, r = tid & 31;
        int gr = blockIdx.x * 32 + r;
        if (gr >= rows) gr = rows - 1;        // clamp; OOB rows never stored
        const SegDesc& s = (seg == 0) ? s0 : (seg == 1) ? s1 : s2;
        const float* p = nullptr;
        if (s.base) { int rr = s.idx ? s.idx[gr] : gr; p = s.base + (size_t)rr * s.width; }
        rowp[seg][r] = p;
    }
    __syncthreads();

    // -------- stage full A tile, branch-free (address selects + 1 load) ----
    const int w0 = s0.width, w01 = s0.width + s1.width;
    for (int r = 0; r < 32; ++r) {
        const float* p0 = rowp[0][r];
        const float* p1 = rowp[1][r];
        const float* p2 = rowp[2][r];
        for (int c = tid; c < Kpad; c += 256) {
            int k = (c < K) ? c : (K - 1);    // clamp into valid range
            const float* p; int ko;
            if (k < w0)       { p = p0; ko = k;       }
            else if (k < w01) { p = p1; ko = k - w0;  }
            else              { p = p2; ko = k - w01; }
            float v = p[ko];
            if (c >= K) v = 0.0f;             // zero K-padding
            atile[r * Kpad + c] = f2bf_bits(v);
        }
    }
    __syncthreads();

    // -------- K loop: 2 WMMA per iteration, no barriers --------
    const unsigned short* arow = atile + ((size_t)((wid >> 2) * 16 + (lane & 15))) * Kpad;
    const int kloA = (lane < 16) ? 0 : 8;     // ISA 16-bit A fragment layout
    const int kofB = (lane < 16) ? 0 : 8;     // ISA 16-bit B fragment layout
    const int colc = (col < Npad) ? col : 0;  // keep idle waves in-bounds
    const unsigned short* wlane = Wp + (size_t)colc * 32 + kofB;
    const size_t wstep = (size_t)Npad * 32;   // stride per 32-K chunk

    v8f acc = {};
    for (int kk = 0; kk < Kpad; kk += 64) {
        const unsigned short* a0p = arow + kk + kloA;
        const unsigned short* wq0 = wlane + (size_t)(kk >> 5) * wstep;
        v8us al = *reinterpret_cast<const v8us*>(a0p);
        v8us ah = *reinterpret_cast<const v8us*>(a0p + 16);
        v8us bl = *reinterpret_cast<const v8us*>(wq0);
        v8us bh = *reinterpret_cast<const v8us*>(wq0 + 16);
        acc = __builtin_amdgcn_wmma_f32_16x16x32_bf16(
                  false, mk16(al, ah), false, mk16(bl, bh), (short)0, acc, false, false);

        const unsigned short* a1p = a0p + 32;
        const unsigned short* wq1 = wq0 + wstep;
        al = *reinterpret_cast<const v8us*>(a1p);
        ah = *reinterpret_cast<const v8us*>(a1p + 16);
        bl = *reinterpret_cast<const v8us*>(wq1);
        bh = *reinterpret_cast<const v8us*>(wq1 + 16);
        acc = __builtin_amdgcn_wmma_f32_16x16x32_bf16(
                  false, mk16(al, ah), false, mk16(bl, bh), (short)0, acc, false, false);
    }

    // -------- epilogue --------
    if (col < Nout) {
        const float bv = bias ? bias[col] : 0.0f;
        #pragma unroll
        for (int v = 0; v < 8; ++v) {
            int m  = v + ((lane < 16) ? 0 : 8);   // C/D VGPR layout
            int gr = rowBase + m;
            if (gr >= rows) continue;
            float val = acc[v] + bv;
            if (act == 1) val = fmaxf(val, 0.0f);
            if (res) val += res[(size_t)gr * Nout + col];
            if (scatter) {
                float aH = alpha[(size_t)gr * 4 + (col >> 6)];     // head = col/64
                atomicAdd(&xacc[(size_t)dstIdx[gr] * 256 + col], val * aH);
            } else {
                out[(size_t)gr * Nout + col] = val;
            }
        }
    }
}

// =========================== small kernels ================================
// Pack fp32 weights [K,Nout] -> bf16 fragment order [Kpad/32][Npad][32],
// zero-padded, so each lane's 16 B-values are 2 contiguous 16B vectors.
__global__ void k_packw(const float* __restrict__ W, unsigned short* __restrict__ Wp,
                        int K, int Nout, int Npad, int nCh) {
    int i = blockIdx.x * blockDim.x + threadIdx.x;
    int total = nCh * Npad * 32;
    if (i >= total) return;
    int j   = i & 31;
    int colc = (i >> 5) % Npad;
    int ch  = (i >> 5) / Npad;
    int k   = ch * 32 + j;
    float v = (k < K && colc < Nout) ? W[(size_t)k * Nout + colc] : 0.0f;
    Wp[i] = f2bf_bits(v);
}
__global__ void k_clean(const float* __restrict__ x_in, const float* __restrict__ dummy,
                        float* __restrict__ xc, int N) {
    int i = blockIdx.x * blockDim.x + threadIdx.x;
    if (i >= N * 16) return;
    int r = i >> 4, c = i & 15;
    bool inv = (x_in[(size_t)r * 16] == -999.0f);
    xc[i] = inv ? dummy[c] : x_in[i];
}
__global__ void k_fill_u32(unsigned* p, unsigned v, int n) {
    int i = blockIdx.x * blockDim.x + threadIdx.x; if (i < n) p[i] = v;
}
__global__ void k_fill_f32(float* p, float v, int n) {
    int i = blockIdx.x * blockDim.x + threadIdx.x; if (i < n) p[i] = v;
}
__global__ void k_segmax(const float* __restrict__ sc, const int* __restrict__ dst,
                         unsigned* __restrict__ smaxU, int E4) {
    int i = blockIdx.x * blockDim.x + threadIdx.x; if (i >= E4) return;
    int e = i >> 2, h = i & 3;
    atomicMax(&smaxU[(size_t)dst[e] * 4 + h], ford_enc(sc[i]));
}
__global__ void k_expden(float* __restrict__ sc, const int* __restrict__ dst,
                         const unsigned* __restrict__ smaxU, float* __restrict__ den, int E4) {
    int i = blockIdx.x * blockDim.x + threadIdx.x; if (i >= E4) return;
    int e = i >> 2, h = i & 3;
    float m  = ford_dec(smaxU[(size_t)dst[e] * 4 + h]);
    float ex = __expf(sc[i] - m);
    sc[i] = ex;
    atomicAdd(&den[(size_t)dst[e] * 4 + h], ex);
}
__global__ void k_alpha(float* __restrict__ sc, const int* __restrict__ dst,
                        const float* __restrict__ den, int E4) {
    int i = blockIdx.x * blockDim.x + threadIdx.x; if (i >= E4) return;
    int e = i >> 2, h = i & 3;
    sc[i] = sc[i] / (den[(size_t)dst[e] * 4 + h] + 1e-16f);
}
// out = maybe_relu(LN(a (+ addend)) * g + b), one block per row, blockDim = D
__global__ void k_res_ln(float* __restrict__ out, const float* __restrict__ a,
                         const float* __restrict__ addend,
                         const float* __restrict__ g, const float* __restrict__ b,
                         int D, int relu) {
    __shared__ float sh[256];
    int row = blockIdx.x, t = threadIdx.x;
    float v = a[(size_t)row * D + t] + (addend ? addend[(size_t)row * D + t] : 0.0f);
    sh[t] = v; __syncthreads();
    for (int s = D >> 1; s > 0; s >>= 1) { if (t < s) sh[t] += sh[t + s]; __syncthreads(); }
    float m = sh[0] / (float)D; __syncthreads();
    float d = v - m;
    sh[t] = d * d; __syncthreads();
    for (int s = D >> 1; s > 0; s >>= 1) { if (t < s) sh[t] += sh[t + s]; __syncthreads(); }
    float var = sh[0] / (float)D;
    float y = d * rsqrtf(var + 1e-5f) * g[t] + b[t];
    if (relu) y = fmaxf(y, 0.0f);
    out[(size_t)row * D + t] = y;
}
__global__ void k_node_softmax(const float* __restrict__ lg, float* __restrict__ pr, int N) {
    int r = blockIdx.x * blockDim.x + threadIdx.x; if (r >= N) return;
    float m = -3.4e38f;
    for (int j = 0; j < 7; ++j) m = fmaxf(m, lg[(size_t)r * 7 + j]);
    float e[7], s = 0.0f;
    for (int j = 0; j < 7; ++j) { e[j] = __expf(lg[(size_t)r * 7 + j] - m); s += e[j]; }
    for (int j = 0; j < 7; ++j) pr[(size_t)r * 7 + j] = e[j] / s;
}
__global__ void k_sigmoid(const float* __restrict__ x, float* __restrict__ y, int n) {
    int i = blockIdx.x * blockDim.x + threadIdx.x;
    if (i < n) y[i] = 1.0f / (1.0f + __expf(-x[i]));
}

// ================================ host ====================================
extern "C" void kernel_launch(void* const* d_in, const int* in_sizes, int n_in,
                              void* d_out, int out_size, void* d_ws, size_t ws_size,
                              hipStream_t stream)
{
    const int N = in_sizes[0] / 16;   // NODE_IN = 16
    const int E = in_sizes[2] / 8;    // EDGE_IN = 8

    const float* x_in  = (const float*)d_in[0];
    const int*   eidx  = (const int*)d_in[1];
    const float* eattr = (const float*)d_in[2];
    const int* src = eidx;            // edge_index[0]
    const int* dst = eidx + E;        // edge_index[1]

    int pi = 3;
    auto F = [&]() -> const float* { return (const float*)d_in[pi++]; };
    const float* p_dummy = F();
    const float *ne_w1 = F(), *ne_b1 = F(), *ne_w2 = F(), *ne_b2 = F();
    const float *ee_w1 = F(), *ee_b1 = F(), *ee_w2 = F(), *ee_b2 = F();
    struct ConvP { const float *aw1,*ab1,*aw2,*ab2,*mw1,*mb1,*mw2,*mb2; } cp[3];
    for (int c = 0; c < 3; ++c) {
        cp[c].aw1=F(); cp[c].ab1=F(); cp[c].aw2=F(); cp[c].ab2=F();
        cp[c].mw1=F(); cp[c].mb1=F(); cp[c].mw2=F(); cp[c].mb2=F();
    }
    struct EuP { const float *w1,*b1,*w2,*b2,*g,*bb; } eu[2];
    for (int c = 0; c < 2; ++c) { eu[c].w1=F(); eu[c].b1=F(); eu[c].w2=F(); eu[c].b2=F(); eu[c].g=F(); eu[c].bb=F(); }
    const float *sk0_w=F(), *sk0_b=F(), *skL_w=F(), *skL_b=F();
    const float *lng[3], *lnb[3];
    for (int c = 0; c < 3; ++c) { lng[c]=F(); lnb[c]=F(); }
    const float *nh_w[4], *nh_b[4];  for (int c=0;c<4;++c){ nh_w[c]=F(); nh_b[c]=F(); }
    const float *eh_w[4], *eh_b[4];  for (int c=0;c<4;++c){ eh_w[c]=F(); eh_b[c]=F(); }

    // ---------------- workspace bump allocator (fp32 units) ----------------
    size_t off = 0;
    auto alloc = [&](size_t n) { float* p = (float*)d_ws + off; off += n; return p; };
    float* x_clean = alloc((size_t)N * 16);
    float* h_n64   = alloc((size_t)N * 64);
    float* h_n128  = alloc((size_t)N * 128);
    float* x0      = alloc((size_t)N * 64);
    float* xacc    = alloc((size_t)N * 256);
    float* x1      = alloc((size_t)N * 256);
    float* x2      = alloc((size_t)N * 256);
    float* x3      = alloc((size_t)N * 256);
    float* e0      = alloc((size_t)E * 64);
    float* e1      = alloc((size_t)E * 64);
    float* e2      = alloc((size_t)E * 64);
    float* e_upd   = alloc((size_t)E * 64);
    float* h_e64   = alloc((size_t)E * 64);
    float* h_e128  = alloc((size_t)E * 128);
    float* h_e256  = alloc((size_t)E * 256);
    float* scores  = alloc((size_t)E * 4);
    unsigned* smaxU = (unsigned*)alloc((size_t)N * 4);
    float* den     = alloc((size_t)N * 4);
    unsigned short* bfpool = (unsigned short*)alloc(800000);  // 1.6M bf16 slots
    if (off * sizeof(float) > ws_size) return;                // scratch too small

    size_t bfoff = 0;
    auto packW = [&](const float* w, int K, int Nout) -> const unsigned short* {
        int Kpad = (K + 63) & ~63;
        int Npad = (Nout + 15) & ~15;
        int nCh  = Kpad / 32;
        size_t n = (size_t)nCh * Npad * 32;
        unsigned short* dp = bfpool + bfoff; bfoff += n;
        k_packw<<<dim3((unsigned)((n + 255) / 256)), 256, 0, stream>>>(w, dp, K, Nout, Npad, nCh);
        return dp;
    };

    const unsigned short *NE1=packW(ne_w1,16,64), *NE2=packW(ne_w2,64,64);
    const unsigned short *EE1=packW(ee_w1,8,64),  *EE2=packW(ee_w2,64,64);
    const unsigned short *CA1[3], *CA2[3], *CM1[3], *CM2[3];
    const int nf[3] = {64, 256, 256};
    for (int c = 0; c < 3; ++c) {
        CA1[c]=packW(cp[c].aw1, 2*nf[c]+64, 64);  CA2[c]=packW(cp[c].aw2, 64, 4);
        CM1[c]=packW(cp[c].mw1, nf[c]+64, 256);   CM2[c]=packW(cp[c].mw2, 256, 256);
    }
    const unsigned short *EU1[2], *EU2[2];
    for (int c = 0; c < 2; ++c) { EU1[c]=packW(eu[c].w1,576,64); EU2[c]=packW(eu[c].w2,64,64); }
    const unsigned short *SK0=packW(sk0_w,64,256), *SKL=packW(skL_w,256,256);
    const unsigned short *NH[4]; const int nhK[5]={256,128,64,32,7};
    for (int c = 0; c < 4; ++c) NH[c]=packW(nh_w[c], nhK[c], nhK[c+1]);
    const unsigned short *EH[4]; const int ehK[5]={576,256,128,64,1};
    for (int c = 0; c < 4; ++c) EH[c]=packW(eh_w[c], ehK[c], ehK[c+1]);

    // ------------------------------ helpers --------------------------------
    auto SD = [](const float* b, const int* ix, int w) { SegDesc s{b, ix, w}; return s; };
    SegDesc Z{nullptr, nullptr, 0};
    auto gemm = [&](SegDesc s0, SegDesc s1, SegDesc s2, int rows, int K,
                    const unsigned short* W, const float* bias, int Nout,
                    float* out, const float* res, int act) {
        dim3 g((rows + 31) / 32, (Nout + 63) / 64);
        k_gemm<<<g, 256, 0, stream>>>(s0, s1, s2, rows, K, W, bias, Nout,
                                      out, res, act, 0, nullptr, nullptr, nullptr);
    };
    auto gemm_scatter = [&](SegDesc s0, int rows, int K,
                            const unsigned short* W, const float* bias) {
        dim3 g((rows + 31) / 32, 4);  // Nout = 256
        k_gemm<<<g, 256, 0, stream>>>(s0, Z, Z, rows, K, W, bias, 256,
                                      nullptr, nullptr, 0, 1, scores, dst, xacc);
    };
    const int E4 = E * 4, N4 = N * 4;
    auto runConv = [&](const float* xin, int nfc, const float* ecur, int c,
                       const float* g, const float* bb, float* xout) {
        // attention scores: [x_dst | x_src | e] -> 64 -> 4
        gemm(SD(xin, dst, nfc), SD(xin, src, nfc), SD(ecur, nullptr, 64),
             E, 2 * nfc + 64, CA1[c], cp[c].ab1, 64, h_e64, nullptr, 1);
        gemm(SD(h_e64, nullptr, 64), Z, Z, E, 64, CA2[c], cp[c].ab2, 4, scores, nullptr, 0);
        // segment softmax over dst
        k_fill_u32<<<dim3((N4+255)/256),256,0,stream>>>(smaxU, 0u, N4);
        k_fill_f32<<<dim3((N4+255)/256),256,0,stream>>>(den, 0.0f, N4);
        k_segmax<<<dim3((E4+255)/256),256,0,stream>>>(scores, dst, smaxU, E4);
        k_expden<<<dim3((E4+255)/256),256,0,stream>>>(scores, dst, smaxU, den, E4);
        k_alpha <<<dim3((E4+255)/256),256,0,stream>>>(scores, dst, den, E4);
        // messages: [x_src | e] -> 256 -> 256, alpha-weighted scatter into xacc
        gemm(SD(xin, src, nfc), SD(ecur, nullptr, 64), Z,
             E, nfc + 64, CM1[c], cp[c].mb1, 256, h_e256, nullptr, 1);
        gemm_scatter(SD(h_e256, nullptr, 256), E, 256, CM2[c], cp[c].mb2);
        // x = relu(LN(xacc))
        k_res_ln<<<dim3((unsigned)N), 256, 0, stream>>>(xout, xacc, nullptr, g, bb, 256, 1);
    };
    auto runEU = [&](const float* xin, const float* ein, int c, float* eout) {
        gemm(SD(xin, dst, 256), SD(xin, src, 256), SD(ein, nullptr, 64),
             E, 576, EU1[c], eu[c].b1, 64, h_e64, nullptr, 1);
        gemm(SD(h_e64, nullptr, 64), Z, Z, E, 64, EU2[c], eu[c].b2, 64, e_upd, nullptr, 0);
        k_res_ln<<<dim3((unsigned)E), 64, 0, stream>>>(eout, ein, e_upd, eu[c].g, eu[c].bb, 64, 0);
    };

    // ------------------------------ pipeline -------------------------------
    k_clean<<<dim3((N*16+255)/256),256,0,stream>>>(x_in, p_dummy, x_clean, N);
    gemm(SD(x_clean, nullptr, 16), Z, Z, N, 16, NE1, ne_b1, 64, h_n64, nullptr, 1);
    gemm(SD(h_n64,  nullptr, 64), Z, Z, N, 64, NE2, ne_b2, 64, x0, nullptr, 0);
    gemm(SD(eattr,  nullptr, 8),  Z, Z, E, 8,  EE1, ee_b1, 64, h_e64, nullptr, 1);
    gemm(SD(h_e64,  nullptr, 64), Z, Z, E, 64, EE2, ee_b2, 64, e0, nullptr, 0);

    // conv1 (skip = linear skip0(x0), written into xacc first)
    gemm(SD(x0, nullptr, 64), Z, Z, N, 64, SK0, sk0_b, 256, xacc, nullptr, 0);
    runConv(x0, 64, e0, 0, lng[0], lnb[0], x1);
    runEU(x1, e0, 0, e1);

    // conv2 (identity skip)
    hipMemcpyAsync(xacc, x1, (size_t)N*256*sizeof(float), hipMemcpyDeviceToDevice, stream);
    runConv(x1, 256, e1, 1, lng[1], lnb[1], x2);
    runEU(x2, e1, 1, e2);

    // conv3 (identity skip)
    hipMemcpyAsync(xacc, x2, (size_t)N*256*sizeof(float), hipMemcpyDeviceToDevice, stream);
    runConv(x2, 256, e2, 2, lng[2], lnb[2], x3);

    // x_out = x3 + skip_long(x1)  (written into xacc)
    gemm(SD(x1, nullptr, 256), Z, Z, N, 256, SKL, skL_b, 256, xacc, x3, 0);
    const float* x_out = xacc;

    // ------------------------------- outputs -------------------------------
    float* out_nl = (float*)d_out;
    float* out_el = out_nl + (size_t)N * 7;
    float* out_np = out_el + (size_t)E;
    float* out_ep = out_np + (size_t)N * 7;

    // node head 256 -> 128 -> 64 -> 32 -> 7
    gemm(SD(x_out, nullptr, 256), Z, Z, N, 256, NH[0], nh_b[0], 128, h_n128, nullptr, 1);
    gemm(SD(h_n128, nullptr, 128), Z, Z, N, 128, NH[1], nh_b[1], 64,  h_n64,  nullptr, 1);
    gemm(SD(h_n64,  nullptr, 64),  Z, Z, N, 64,  NH[2], nh_b[2], 32,  h_n128, nullptr, 1);
    gemm(SD(h_n128, nullptr, 32),  Z, Z, N, 32,  NH[3], nh_b[3], 7,   out_nl, nullptr, 0);
    k_node_softmax<<<dim3((N+255)/256),256,0,stream>>>(out_nl, out_np, N);

    // edge head [x_out[src] | x_out[dst] | e2] : 576 -> 256 -> 128 -> 64 -> 1
    gemm(SD(x_out, src, 256), SD(x_out, dst, 256), SD(e2, nullptr, 64),
         E, 576, EH[0], eh_b[0], 256, h_e256, nullptr, 1);
    gemm(SD(h_e256, nullptr, 256), Z, Z, E, 256, EH[1], eh_b[1], 128, h_e128, nullptr, 1);
    gemm(SD(h_e128, nullptr, 128), Z, Z, E, 128, EH[2], eh_b[2], 64,  h_e64,  nullptr, 1);
    gemm(SD(h_e64,  nullptr, 64),  Z, Z, E, 64,  EH[3], eh_b[3], 1,   out_el, nullptr, 0);
    k_sigmoid<<<dim3((E+255)/256),256,0,stream>>>(out_el, out_ep, E);
}
